// TransformerBlock_52140902973586
// MI455X (gfx1250) — matrix operand order repaired and checked by reading, not compiled, not run on hardware
//
#include <hip/hip_runtime.h>

// ---------------------------------------------------------------------------
// Types for CDNA5 WMMA (wave32): v_wmma_f32_16x16x32_bf16
// ---------------------------------------------------------------------------
typedef __attribute__((ext_vector_type(8)))  __bf16 v8bf;
typedef __attribute__((ext_vector_type(16))) __bf16 v16bf;
typedef __attribute__((ext_vector_type(8)))  float  v8f;

#define BB 8
#define CC 1024
#define HH 64
#define WW 64
#define HWC 4096          // H*W
#define NHH 16
#define HD 64
#define EPS 1e-3f

static __device__ __forceinline__ v16bf ld_frag(const __bf16* p0, const __bf16* p1) {
  // Two 16-byte LDS reads -> one 32-byte fragment (8 VGPRs of packed bf16).
  v8bf lo = *(const v8bf*)p0;
  v8bf hi = *(const v8bf*)p1;
  v16bf r;
#pragma unroll
  for (int i = 0; i < 8; ++i) { r[i] = lo[i]; r[8 + i] = hi[i]; }
  return r;
}

static __device__ __forceinline__ float silu(float y) {
  return y / (1.0f + __expf(-y));
}

// 16 consecutive elements -> f32, via guaranteed 16B vector loads.
static __device__ __forceinline__ void load16f(const float* src, float* o) {
  const float4* s = (const float4*)src;
  float4 a = s[0], b = s[1], c = s[2], d = s[3];
  o[0]=a.x; o[1]=a.y; o[2]=a.z;  o[3]=a.w;
  o[4]=b.x; o[5]=b.y; o[6]=b.z;  o[7]=b.w;
  o[8]=c.x; o[9]=c.y; o[10]=c.z; o[11]=c.w;
  o[12]=d.x; o[13]=d.y; o[14]=d.z; o[15]=d.w;
}
static __device__ __forceinline__ void load16f(const __bf16* src, float* o) {
  const v8bf* s = (const v8bf*)src;
  v8bf a = s[0], b = s[1];
#pragma unroll
  for (int i = 0; i < 8; ++i) { o[i] = (float)a[i]; o[8 + i] = (float)b[i]; }
}
static __device__ __forceinline__ void store16bf(__bf16* dst, const float* o) {
  v8bf h0, h1;
#pragma unroll
  for (int i = 0; i < 8; ++i) { h0[i] = (__bf16)o[i]; h1[i] = (__bf16)o[8 + i]; }
  *(v8bf*)dst = h0;
  *(v8bf*)(dst + 8) = h1;
}

// ---------------------------------------------------------------------------
// Kernel 1/4: 1x1 conv as GEMM  Y[b,co,hw] = sum_ci W[co,ci] * X[b,ci,hw]
// Block tile 128(M) x 128(N), K-step 32, DOUBLE-BUFFERED LDS: the k+32
// global loads issue before computing on the current buffer, so HBM/L2
// latency overlaps the 8 v_wmma_f32_16x16x32_bf16 per wave per K-step.
// Fused BN+SiLU epilogue; optional f32 shortcut add (conv3).
// ---------------------------------------------------------------------------
template <typename TIn, bool ADD_SHORTCUT, typename TOut>
__global__ __launch_bounds__(256) void gemm1x1_bn_silu(
    const float* __restrict__ Wmat,      // [C][C] f32 weights
    const TIn*   __restrict__ Xin,       // [B][C][HW]
    const float* __restrict__ g,  const float* __restrict__ bta,
    const float* __restrict__ mu, const float* __restrict__ var,
    const float* __restrict__ shortcut,  // [B][C][HW] f32 (conv3 only)
    TOut*        __restrict__ Yout)      // [B][C][HW]
{
  __shared__ __bf16 As[2][128][40];  // [m][k], rows 80B -> 16B aligned chunks
  __shared__ __bf16 Bs[2][128][40];  // [n][k] (transposed stage, contiguous K)

  const int t    = threadIdx.x;
  const int lane = t & 31;
  const int wv   = t >> 5;          // 0..7
  const int wm   = wv & 3;          // M position: 4 waves * 32 rows
  const int wn   = wv >> 2;         // N position: 2 waves * 64 cols
  const int half = lane >> 4;       // K-half select per ISA layout
  const int lr   = lane & 15;

  const int b   = blockIdx.z;
  const int co0 = blockIdx.y * 128;
  const int hw0 = blockIdx.x * 128;
  const TIn* Xb = Xin + (size_t)b * CC * HWC;

  v8f acc[2][4];
#pragma unroll
  for (int i = 0; i < 2; ++i)
#pragma unroll
    for (int j = 0; j < 4; ++j) acc[i][j] = (v8f)(0.0f);

  const int ar = t >> 1, ac = (t & 1) * 16;  // A stage: 128 rows x 32 cols
  const int br = t >> 3, bc = (t & 7) * 16;  // B stage: 32 rows x 128 cols

  auto stageA = [&](int k0, int buf) {
    float tmp[16];
    load16f(Wmat + (size_t)(co0 + ar) * CC + k0 + ac, tmp);
    store16bf(&As[buf][ar][ac], tmp);          // two 16B LDS stores
  };
  auto stageB = [&](int k0, int buf) {
    float tmp[16];
    load16f(Xb + (size_t)(k0 + br) * HWC + hw0 + bc, tmp);
#pragma unroll
    for (int i = 0; i < 16; ++i) Bs[buf][bc + i][br] = (__bf16)tmp[i];
  };

  stageA(0, 0);
  stageB(0, 0);
  __syncthreads();

  for (int k0 = 0; k0 < CC; k0 += 32) {
    const int cur = (k0 >> 5) & 1;
    const int nxt = cur ^ 1;
    if (k0 + 32 < CC) {           // issue next slab's global loads first
      stageA(k0 + 32, nxt);
      stageB(k0 + 32, nxt);
      __builtin_prefetch(Wmat + (size_t)(co0 + ar) * CC + k0 + 64 + ac, 0, 1);
      __builtin_prefetch(Xb + (size_t)(k0 + 64 + br) * HWC + hw0 + bc, 0, 1);
    }

    v16bf af[2], bf[4];
#pragma unroll
    for (int mt = 0; mt < 2; ++mt) {
      const __bf16* pa = &As[cur][wm * 32 + mt * 16 + lr][0];
      af[mt] = ld_frag(pa + half * 8, pa + 16 + half * 8);
    }
#pragma unroll
    for (int nt = 0; nt < 4; ++nt) {
      const __bf16* pb = &Bs[cur][wn * 64 + nt * 16 + lr][0];
      bf[nt] = ld_frag(pb + half * 8, pb + 16 + half * 8);
    }
#pragma unroll
    for (int mt = 0; mt < 2; ++mt)
#pragma unroll
      for (int nt = 0; nt < 4; ++nt)
        acc[mt][nt] = __builtin_amdgcn_wmma_f32_16x16x32_bf16(
            false, af[mt], false, bf[nt], (short)0, acc[mt][nt], false, false);
    __syncthreads();
  }

  // ---- epilogue: BN + SiLU (+ shortcut), C/D layout per ISA 7.12.2 ----
#pragma unroll
  for (int mt = 0; mt < 2; ++mt) {
#pragma unroll
    for (int r = 0; r < 8; ++r) {
      const int co = co0 + wm * 32 + mt * 16 + r + half * 8;
      const float sc = g[co] * rsqrtf(var[co] + EPS);
      const float bi = bta[co] - mu[co] * sc;
#pragma unroll
      for (int nt = 0; nt < 4; ++nt) {
        const int hw = hw0 + wn * 64 + nt * 16 + lr;
        float y = silu(acc[mt][nt][r] * sc + bi);
        const size_t oidx = ((size_t)b * CC + co) * HWC + hw;
        if (ADD_SHORTCUT) y += shortcut[oidx];
        Yout[oidx] = (TOut)y;
      }
    }
  }
}

// ---------------------------------------------------------------------------
// Kernel 2: per-(b,nh,h) channel softmax + att = P @ A^T (64x64x64) via WMMA.
// A[w][hd] loaded to LDS in f32, softmax over hd, P/A re-staged as bf16,
// 16 output tiles of 16x16 -> 2 per wave, 2 WMMA each (K=64).
// Result written in permuted layout: channel = nh*64 + v, contiguous in w.
// ---------------------------------------------------------------------------
__global__ __launch_bounds__(256) void attn_kernel(
    const __bf16* __restrict__ Y1, __bf16* __restrict__ Y2)
{
  __shared__ float  Af[64][65];
  __shared__ __bf16 Pl[64][72];
  __shared__ __bf16 Al[64][72];

  const int t  = threadIdx.x;
  const int h  = blockIdx.x;
  const int nh = blockIdx.y;
  const int b  = blockIdx.z;
  const size_t base = (((size_t)b * NHH + nh) * HD) * HWC + (size_t)h * WW;

  // ---- load a[w][hd] (two 16B vector loads per thread) ----
  {
    const int hd = t >> 2;
    const int w0 = (t & 3) * 16;
    float tmp[16];
    load16f(Y1 + base + (size_t)hd * HWC + w0, tmp);
#pragma unroll
    for (int i = 0; i < 16; ++i) Af[w0 + i][hd] = tmp[i];
  }
  __syncthreads();

  // ---- rowwise softmax over hd (64 rows) ----
  if (t < 64) {
    float mx = -3.0e38f;
    for (int k = 0; k < 64; ++k) mx = fmaxf(mx, Af[t][k]);
    float s = 0.0f;
    for (int k = 0; k < 64; ++k) s += __expf(Af[t][k] - mx);
    const float inv = 1.0f / s;
    for (int k = 0; k < 64; ++k) {
      const float v = Af[t][k];
      Pl[t][k] = (__bf16)(__expf(v - mx) * inv);
      Al[t][k] = (__bf16)v;
    }
  }
  __syncthreads();

  // ---- att = P (64x64) x A^T : B-fragment cols are rows of A ----
  const int lane = t & 31, wv = t >> 5;
  const int half = lane >> 4, lr = lane & 15;
#pragma unroll
  for (int ti = 0; ti < 2; ++ti) {
    const int T  = wv * 2 + ti;
    const int tm = T & 3, tn = T >> 2;
    v8f acc = (v8f)(0.0f);
#pragma unroll
    for (int k0 = 0; k0 < 64; k0 += 32) {
      const __bf16* pa = &Pl[tm * 16 + lr][k0];
      v16bf af = ld_frag(pa + half * 8, pa + 16 + half * 8);
      const __bf16* pb = &Al[tn * 16 + lr][k0];
      v16bf bf = ld_frag(pb + half * 8, pb + 16 + half * 8);
      acc = __builtin_amdgcn_wmma_f32_16x16x32_bf16(
          false, af, false, bf, (short)0, acc, false, false);
    }
    // transpose through LDS: att[v][w]
#pragma unroll
    for (int r = 0; r < 8; ++r) {
      const int w = tm * 16 + r + half * 8;
      const int v = tn * 16 + lr;
      Af[v][w] = acc[r];
    }
  }
  __syncthreads();

  // ---- coalesced writeback: Y2[b, nh*64+v, h, w] (two 16B stores) ----
  {
    const int v  = t >> 2;
    const int w0 = (t & 3) * 16;
    float tmp[16];
#pragma unroll
    for (int i = 0; i < 16; ++i) tmp[i] = Af[v][w0 + i];
    store16bf(Y2 + base + (size_t)v * HWC + w0, tmp);
  }
}

// ---------------------------------------------------------------------------
// Kernel 3: depthwise 3x3 conv (SAME) + BN + SiLU. Memory bound; whole
// 64 MiB bf16 tensor fits in the 192 MB L2, so 9x re-reads mostly hit.
// ---------------------------------------------------------------------------
__global__ __launch_bounds__(256) void dwconv_bn_silu(
    const __bf16* __restrict__ Yin, const float* __restrict__ W2,
    const float* __restrict__ g,  const float* __restrict__ bta,
    const float* __restrict__ mu, const float* __restrict__ var,
    __bf16* __restrict__ Yout)
{
  const size_t idx = (size_t)blockIdx.x * 256 + threadIdx.x;  // B*C*H*W
  const int w = (int)(idx & 63);
  const int h = (int)((idx >> 6) & 63);
  const int c = (int)((idx >> 12) & 1023);
  const __bf16* plane = Yin + (idx & ~(size_t)(HWC - 1));
  const float* wk = W2 + c * 9;

  float sum = 0.0f;
#pragma unroll
  for (int dy = -1; dy <= 1; ++dy) {
    const int hh = h + dy;
    if ((unsigned)hh < (unsigned)HH) {
#pragma unroll
      for (int dx = -1; dx <= 1; ++dx) {
        const int ww = w + dx;
        if ((unsigned)ww < (unsigned)WW)
          sum += wk[(dy + 1) * 3 + (dx + 1)] * (float)plane[hh * WW + ww];
      }
    }
  }
  const float sc = g[c] * rsqrtf(var[c] + EPS);
  const float bi = bta[c] - mu[c] * sc;
  Yout[idx] = (__bf16)silu(sum * sc + bi);
}

// ---------------------------------------------------------------------------
// Launch: conv1(WMMA) -> attention(WMMA) -> dw3x3 -> conv3(WMMA)+shortcut
// Workspace: y1 (64 MiB bf16) | y2 (64 MiB bf16); dw output reuses y1's slot.
// ---------------------------------------------------------------------------
extern "C" void kernel_launch(void* const* d_in, const int* in_sizes, int n_in,
                              void* d_out, int out_size, void* d_ws, size_t ws_size,
                              hipStream_t stream) {
  const float* x  = (const float*)d_in[0];
  const float* w1 = (const float*)d_in[1];
  const float* w2 = (const float*)d_in[2];
  const float* w3 = (const float*)d_in[3];
  const float* g1 = (const float*)d_in[4];
  const float* b1 = (const float*)d_in[5];
  const float* m1 = (const float*)d_in[6];
  const float* v1 = (const float*)d_in[7];
  const float* g2 = (const float*)d_in[8];
  const float* b2 = (const float*)d_in[9];
  const float* m2 = (const float*)d_in[10];
  const float* v2 = (const float*)d_in[11];
  const float* g3 = (const float*)d_in[12];
  const float* b3 = (const float*)d_in[13];
  const float* m3 = (const float*)d_in[14];
  const float* v3 = (const float*)d_in[15];

  const size_t nelem = (size_t)BB * CC * HWC;  // 33,554,432
  __bf16* y1 = (__bf16*)d_ws;
  __bf16* y2 = y1 + nelem;
  __bf16* y3 = y1;  // reuse y1's slot after attention consumed it
  float*  out = (float*)d_out;

  const dim3 blk(256);
  const dim3 gemm_grid(HWC / 128, CC / 128, BB);  // (32, 8, 8)

  gemm1x1_bn_silu<float, false, __bf16><<<gemm_grid, blk, 0, stream>>>(
      w1, x, g1, b1, m1, v1, nullptr, y1);

  attn_kernel<<<dim3(HH, NHH, BB), blk, 0, stream>>>(y1, y2);

  dwconv_bn_silu<<<dim3((unsigned)(nelem / 256)), blk, 0, stream>>>(
      y2, w2, g2, b2, m2, v2, y3);

  gemm1x1_bn_silu<__bf16, true, float><<<gemm_grid, blk, 0, stream>>>(
      w3, y3, g3, b3, m3, v3, x, out);
}
